// Hgru2_1743756722694
// MI455X (gfx1250) — matrix operand order repaired and checked by reading, not compiled, not run on hardware
//
#include <hip/hip_runtime.h>
#include <hip/hip_bf16.h>
#include <cstdint>
#include <cstddef>

typedef __attribute__((ext_vector_type(16))) __bf16 v16bf;
typedef __attribute__((ext_vector_type(8)))  __bf16 v8bf;
typedef __attribute__((ext_vector_type(4)))  __bf16 v4bf;
typedef __attribute__((ext_vector_type(2)))  __bf16 v2bf;
typedef __attribute__((ext_vector_type(8)))  float  v8f;
typedef int v4i __attribute__((vector_size(16)));   // matches builtin param pointee

#define AS1 __attribute__((address_space(1)))
#define AS3 __attribute__((address_space(3)))

// gfx1250 async global->LDS copy (ASYNCcnt path); falls back to ld/st staging.
#if __has_builtin(__builtin_amdgcn_global_load_async_to_lds_b128)
#define ASYNC_COPY 1
#else
#define ASYNC_COPY 0
#endif

__device__ __forceinline__ void wait_async_zero() {
#if ASYNC_COPY
#if __has_builtin(__builtin_amdgcn_s_wait_asynccnt)
    __builtin_amdgcn_s_wait_asynccnt(0);
#else
    asm volatile("s_wait_asynccnt 0x0" ::: "memory");
#endif
#endif
}

// ---------------- problem constants ----------------
#define NSEQ 4096
#define BATCH 8
#define DMODEL 1024
#define MROWS (NSEQ * BATCH)   // 32768

// ---------------- GEMM tiling ----------------
#define BM 256                 // block tile M (8 waves, each 64x64)
#define BN 128                 // block tile N
#define BK 32                  // one bf16-WMMA K step
#define LDSS 40                // bf16 elems per LDS row: 32 data + 8 pad = 80B = 5*16B
                               // -> every 16B chunk aligned; A-frag reads conflict-free

// convert f32 -> bf16, 4 elems/thread
__global__ void cvt_x_kernel(const float* __restrict__ x, __bf16* __restrict__ xb, int count) {
    int i = (blockIdx.x * blockDim.x + threadIdx.x) * 4;
    if (i >= count) return;
    float4 f = *(const float4*)(x + i);
    v4bf o;
    o.x = (__bf16)f.x; o.y = (__bf16)f.y; o.z = (__bf16)f.z; o.w = (__bf16)f.w;
    *(v4bf*)(xb + i) = o;
}

// W: K x N row-major (f32)  ->  Wt: N x K row-major (bf16)  (B^T so B frags are contiguous)
__global__ void cvt_transpose_kernel(const float* __restrict__ W, __bf16* __restrict__ Wt,
                                     int K, int N) {
    int idx = blockIdx.x * blockDim.x + threadIdx.x;
    if (idx >= K * N) return;
    int nIdx = idx / K;
    int k    = idx - nIdx * K;
    Wt[idx] = (__bf16)W[(size_t)k * N + nIdx];
}

// D = A(bf16, MxK row-major) x Bt^T, Bt is (N x K) bf16 row-major.
// mode 0: Cf[row*N + col] = acc (f32)
// mode 1: col < 2d  -> Cvq[row*2d + col] = gelu(acc)                  (bf16)
//         col >= 2d -> Clam[row*d + (col-2d)] = lb + (1-lb)*sigmoid   (f32)
__global__ __launch_bounds__(256)
void gemm_wmma_kernel(const __bf16* __restrict__ A, const __bf16* __restrict__ Bt,
                      float* __restrict__ Cf,
                      __bf16* __restrict__ Cvq, float* __restrict__ Clam,
                      const float* __restrict__ lb,
                      int M, int N, int K, int mode) {
    __shared__ __align__(16) __bf16 As[2][BM * LDSS];   // 2*256*40*2 = 40 KB
    __shared__ __align__(16) __bf16 Bs[2][BN * LDSS];   // 2*128*40*2 = 20 KB

    const int tid  = threadIdx.x;
    const int wave = tid >> 5;          // 8 waves (wave32)
    const int lane = tid & 31;
    const int wm   = wave >> 1;         // 0..3 : 64-row strip
    const int wn   = wave & 1;          // 0..1 : 64-col strip
    const int hlf  = lane >> 4;         // lane half per ISA frag layout
    const int l16  = lane & 15;

    const int blockM = blockIdx.y * BM;
    const int blockN = blockIdx.x * BN;

    const v8f vzero = {0.f, 0.f, 0.f, 0.f, 0.f, 0.f, 0.f, 0.f};
    v8f acc[4][4];
#pragma unroll
    for (int i = 0; i < 4; ++i)
#pragma unroll
        for (int j = 0; j < 4; ++j) acc[i][j] = vzero;

    // stage one BK-slice of A (BM x BK) and Bt (BN x BK) into LDS buffer `buf`
    auto stage = [&](int k0, int buf) {
#if ASYNC_COPY
        // direct global->LDS DMA, no VGPR round trip, tracked by ASYNCcnt
#pragma unroll
        for (int it = 0; it < 4; ++it) {                    // A: 1024 chunks of 8 bf16
            int c = tid + it * 256, row = c >> 2, kc = (c & 3) * 8;
            __builtin_amdgcn_global_load_async_to_lds_b128(
                (AS1 v4i*)(A + (size_t)(blockM + row) * K + k0 + kc),
                (AS3 v4i*)&As[buf][row * LDSS + kc], 0, 0);
        }
#pragma unroll
        for (int it = 0; it < 2; ++it) {                    // B: 512 chunks
            int c = tid + it * 256, row = c >> 2, kc = (c & 3) * 8;
            __builtin_amdgcn_global_load_async_to_lds_b128(
                (AS1 v4i*)(Bt + (size_t)(blockN + row) * K + k0 + kc),
                (AS3 v4i*)&Bs[buf][row * LDSS + kc], 0, 0);
        }
#else
        // fallback: load ALL chunks into distinct temps first, then store
        int4 ta[4], tb[2];
#pragma unroll
        for (int it = 0; it < 4; ++it) {
            int c = tid + it * 256, row = c >> 2, kc = (c & 3) * 8;
            ta[it] = *(const int4*)(A + (size_t)(blockM + row) * K + k0 + kc);
        }
#pragma unroll
        for (int it = 0; it < 2; ++it) {
            int c = tid + it * 256, row = c >> 2, kc = (c & 3) * 8;
            tb[it] = *(const int4*)(Bt + (size_t)(blockN + row) * K + k0 + kc);
        }
#pragma unroll
        for (int it = 0; it < 4; ++it) {
            int c = tid + it * 256, row = c >> 2, kc = (c & 3) * 8;
            *(int4*)&As[buf][row * LDSS + kc] = ta[it];
        }
#pragma unroll
        for (int it = 0; it < 2; ++it) {
            int c = tid + it * 256, row = c >> 2, kc = (c & 3) * 8;
            *(int4*)&Bs[buf][row * LDSS + kc] = tb[it];
        }
#endif
    };

    stage(0, 0);
    const int nk = K / BK;

    for (int ki = 0; ki < nk; ++ki) {
        wait_async_zero();                                  // our async stores to LDS done
        __syncthreads();                                    // buf(ki) visible; buf(ki+1) free
        if (ki + 1 < nk) stage((ki + 1) * BK, (ki + 1) & 1);
        if (ki + 2 < nk) {                                  // token prefetch of tile ki+2
            __builtin_prefetch(A  + (size_t)(blockM + (tid >> 2)) * K + (ki + 2) * BK, 0, 1);
            __builtin_prefetch(Bt + (size_t)(blockN + (tid >> 3)) * K + (ki + 2) * BK, 0, 1);
        }
        const __bf16* as = As[ki & 1];
        const __bf16* bs = Bs[ki & 1];

        // A frag (16x32): lane<16: K {0..7,16..23}; lane>=16: K {8..15,24..31}
        v16bf afrag[4];
#pragma unroll
        for (int i = 0; i < 4; ++i) {
            int row = wm * 64 + i * 16 + l16;
            v8bf lo = *(const v8bf*)&as[row * LDSS + hlf * 8];
            v8bf hi = *(const v8bf*)&as[row * LDSS + 16 + hlf * 8];
            afrag[i] = __builtin_shufflevector(lo, hi, 0,1,2,3,4,5,6,7,8,9,10,11,12,13,14,15);
        }
        // B frag (32x16): lane<16: K {0..7,16..23}; lane>=16: K {8..15,24..31}, N = l16
        v16bf bfrag[4];
#pragma unroll
        for (int j = 0; j < 4; ++j) {
            int rowb = wn * 64 + j * 16 + l16;
            v8bf lo = *(const v8bf*)&bs[rowb * LDSS + hlf * 8];
            v8bf hi = *(const v8bf*)&bs[rowb * LDSS + 16 + hlf * 8];
            bfrag[j] = __builtin_shufflevector(lo, hi, 0,1,2,3,4,5,6,7,8,9,10,11,12,13,14,15);
        }

#pragma unroll
        for (int i = 0; i < 4; ++i)
#pragma unroll
            for (int j = 0; j < 4; ++j)
                acc[i][j] = __builtin_amdgcn_wmma_f32_16x16x32_bf16(
                    false, afrag[i], false, bfrag[j], (short)0, acc[i][j], false, false);
    }

    // ---- epilogue: C/D layout: VGPR r -> M = r + 8*hlf, N = l16 ----
    const int twoD = 2 * DMODEL;
#pragma unroll
    for (int i = 0; i < 4; ++i) {
#pragma unroll
        for (int j = 0; j < 4; ++j) {
            int col = blockN + wn * 64 + j * 16 + l16;
#pragma unroll
            for (int r = 0; r < 8; ++r) {
                int row = blockM + wm * 64 + i * 16 + hlf * 8 + r;
                float v = acc[i][j][r];
                if (mode == 0) {
                    Cf[(size_t)row * N + col] = v;
                } else {
                    if (col < twoD) {
                        // exact-erf GELU (matches jax.nn.gelu(approximate=False))
                        float g = 0.5f * v * (1.0f + erff(v * 0.70710678118654752f));
                        Cvq[(size_t)row * twoD + col] = (__bf16)g;
                    } else {
                        int hc = col - twoD;
                        float s = 1.0f / (1.0f + __expf(-v));
                        float l = lb[hc];
                        Clam[(size_t)row * DMODEL + hc] = l + (1.0f - l) * s;
                    }
                }
            }
        }
    }
}

// GLA scan: one lane per (b, h) chain; 2x2 f32 state in registers.
// VQ: (M x 2d) bf16 [V | Q], Lam: (M x d) f32, O: (M x d) bf16
__global__ void scan_kernel(const __bf16* __restrict__ VQ, const float* __restrict__ Lam,
                            __bf16* __restrict__ O) {
    int c  = blockIdx.x * blockDim.x + threadIdx.x;   // 0..4095
    int bb = c >> 9;                                  // batch
    int h  = c & 511;                                 // head
    float S00 = 0.f, S01 = 0.f, S10 = 0.f, S11 = 0.f;
    for (int t = 0; t < NSEQ; ++t) {
        size_t row = (size_t)t * BATCH + bb;
        v2bf vv = *(const v2bf*)(VQ + row * (2 * DMODEL) + 2 * h);
        v2bf qq = *(const v2bf*)(VQ + row * (2 * DMODEL) + DMODEL + 2 * h);
        float2 lm = *(const float2*)(Lam + row * DMODEL + 2 * h);
        float v0 = (float)vv.x, v1 = (float)vv.y;
        float q0 = (float)qq.x, q1 = (float)qq.y;
        float k0f = 1.0f - lm.x, k1f = 1.0f - lm.y;
        // S[kk][vv] = lam[kk]*S + k[kk]*v[vv] ; o[vv] = q[0]*S[0][vv] + q[1]*S[1][vv]
        S00 = fmaf(lm.x, S00, k0f * v0);
        S01 = fmaf(lm.x, S01, k0f * v1);
        S10 = fmaf(lm.y, S10, k1f * v0);
        S11 = fmaf(lm.y, S11, k1f * v1);
        float o0 = fmaf(q0, S00, q1 * S10);
        float o1 = fmaf(q0, S01, q1 * S11);
        v2bf o; o.x = (__bf16)o0; o.y = (__bf16)o1;
        *(v2bf*)(O + row * DMODEL + 2 * h) = o;
    }
}

extern "C" void kernel_launch(void* const* d_in, const int* in_sizes, int n_in,
                              void* d_out, int out_size, void* d_ws, size_t ws_size,
                              hipStream_t stream) {
    const float* x    = (const float*)d_in[0];
    const float* lb   = (const float*)d_in[1];
    const float* Win  = (const float*)d_in[2];
    const float* Wout = (const float*)d_in[3];
    float* out = (float*)d_out;

    const int d = DMODEL;
    const int M = MROWS;

    char* ws = (char*)d_ws;
    size_t off = 0;
    __bf16* xb    = (__bf16*)(ws + off); off += (size_t)M * d * 2;        //  64 MB
    __bf16* WinT  = (__bf16*)(ws + off); off += (size_t)3 * d * d * 2;    //   6 MB
    __bf16* WoutT = (__bf16*)(ws + off); off += (size_t)d * d * 2;        //   2 MB
    __bf16* VQ    = (__bf16*)(ws + off); off += (size_t)M * 2 * d * 2;    // 128 MB
    float*  Lam   = (float*) (ws + off); off += (size_t)M * d * 4;        // 128 MB
    __bf16* Ob    = (__bf16*)(ws + off); off += (size_t)M * d * 2;        //  64 MB
    (void)ws_size; (void)in_sizes; (void)n_in; (void)out_size;

    // 1) precision conversion / weight transpose
    cvt_x_kernel<<<(M * d / 4 + 255) / 256, 256, 0, stream>>>(x, xb, M * d);
    cvt_transpose_kernel<<<(3 * d * d + 255) / 256, 256, 0, stream>>>(Win, WinT, d, 3 * d);
    cvt_transpose_kernel<<<(d * d + 255) / 256, 256, 0, stream>>>(Wout, WoutT, d, d);

    // 2) feat = x @ W_in with fused gelu/sigmoid-gate epilogue
    gemm_wmma_kernel<<<dim3(3 * d / BN, M / BM), 256, 0, stream>>>(
        xb, WinT, nullptr, VQ, Lam, lb, M, 3 * d, d, 1);

    // 3) GLA recurrence (4096 independent chains, one lane each)
    scan_kernel<<<(BATCH * (d / 2)) / 256, 256, 0, stream>>>(VQ, Lam, Ob);

    // 4) out = o @ W_out
    gemm_wmma_kernel<<<dim3(d / BN, M / BM), 256, 0, stream>>>(
        Ob, WoutT, out, nullptr, nullptr, nullptr, M, d, d, 0);
}
